// InnerIteration_60936995996046
// MI455X (gfx1250) — matrix operand order repaired and checked by reading, not compiled, not run on hardware
//
#include <hip/hip_runtime.h>
#include <stdint.h>

// ---------------------------------------------------------------------------
// Types for CDNA5 WMMA (gfx1250, wave32)
// ---------------------------------------------------------------------------
typedef __attribute__((ext_vector_type(16))) __bf16 bf16x16;
typedef __attribute__((ext_vector_type(8)))  float  f32x8;

union FragB16 {            // 16 bf16 elements = 8 dwords
    bf16x16  v;
    uint32_t u[8];
};

static __device__ __forceinline__ f32x8 wmma_bf16(const FragB16& a, const FragB16& b, f32x8 c) {
    // D = A(16x32 bf16) * B(32x16 bf16) + C(16x16 f32)
    return __builtin_amdgcn_wmma_f32_16x16x32_bf16(
        /*neg_a=*/false, a.v, /*neg_b=*/false, b.v,
        /*c_mod=*/(short)0, c, /*reuse_a=*/false, /*reuse_b=*/false);
}

// ---------------------------------------------------------------------------
// bf16 helpers (round-to-nearest-even)
// ---------------------------------------------------------------------------
static __device__ __forceinline__ unsigned short f2bf(float f) {
    unsigned u = __float_as_uint(f);
    unsigned r = u + 0x7FFFu + ((u >> 16) & 1u);
    return (unsigned short)(r >> 16);
}
static __device__ __forceinline__ uint32_t pack_bf(float lo, float hi) {
    return (uint32_t)f2bf(lo) | ((uint32_t)f2bf(hi) << 16);
}
static __device__ __forceinline__ float bf2f(unsigned short h) {
    return __uint_as_float(((unsigned)h) << 16);
}
static __device__ __forceinline__ float sigmoidf_(float x) {
    return 1.0f / (1.0f + __expf(-x));
}

// ---------------------------------------------------------------------------
// Fragment loaders.
// A fragment (LDS), row-major [M][Kstride] bf16 stored as packed dwords.
// Layout: lane L holds M=L&15; lane-half selects K offset 0/16; VGPR j
// holds K = k0+koff+2j(+1) (matches B fragment order).
// ---------------------------------------------------------------------------
static __device__ __forceinline__ void load_a_lds(FragB16& f, const uint32_t* smem,
                                                  int rowBase, int strideDW, int k0, int lane) {
    int m  = lane & 15;
    int h4 = (lane >> 4) * 8;                       // 0 or 8 dwords (K offset 0/16)
    int base = (rowBase + m) * strideDW + (k0 >> 1) + h4;
#pragma unroll
    for (int j = 0; j < 8; ++j) f.u[j]     = smem[base + j];
}

// B fragment, pair-interleaved [K/2][64] dwords (element (k,n) in dword
// (k>>1)*64+n, half k&1). Lane holds N=lane&15; lane-half => K offset 16.
// Pass base pre-offset by the 16-column n-tile. Works for global or LDS ptr.
static __device__ __forceinline__ void load_b_glb(FragB16& f, const uint32_t* __restrict__ g,
                                                  int k0, int lane) {
    int n  = lane & 15;
    int kb = (k0 >> 1) + (lane >> 4) * 8;
#pragma unroll
    for (int j = 0; j < 8; ++j) f.u[j] = g[(size_t)(kb + j) * 64 + n];
}
static __device__ __forceinline__ void load_b_lds(FragB16& f, const uint32_t* smem,
                                                  int k0, int lane) {
    int n  = lane & 15;
    int kb = (k0 >> 1) + (lane >> 4) * 8;
#pragma unroll
    for (int j = 0; j < 8; ++j) f.u[j] = smem[(kb + j) * 64 + n];
}

// ---------------------------------------------------------------------------
// Kernel 0: convert a [K][64] fp32 weight to bf16 pair-interleaved dwords.
// ---------------------------------------------------------------------------
__global__ void convert_pairs(const float* __restrict__ src, uint32_t* __restrict__ dst, int ndw) {
    int i = blockIdx.x * blockDim.x + threadIdx.x;
    if (i >= ndw) return;
    int k2 = i >> 6, n = i & 63;
    float lo = src[(size_t)(k2 * 2)     * 64 + n];
    float hi = src[(size_t)(k2 * 2 + 1) * 64 + n];
    dst[i] = pack_bf(lo, hi);
}

// ---------------------------------------------------------------------------
// Fused kernel: per block, 16 variables end-to-end.
// ---------------------------------------------------------------------------
#define NV 16384
#define NC 16
#define NL 6
#define ND 64

__global__ __launch_bounds__(256) void sat_fused_kernel(
    const float* __restrict__ variables,      // [V,64]
    const int* __restrict__ lits,             // [V,16,6]
    const unsigned char* __restrict__ neg_mask,     // [V,16,6] bool
    const unsigned char* __restrict__ var_valid,    // [V,16,6] bool
    const unsigned char* __restrict__ clause_valid, // [V,16]   bool
    const float* __restrict__ Wn,             // [64,64] fp32 (for true_emb)
    const float* __restrict__ bn,             // [64]
    const float* __restrict__ false_emb,      // [64]
    const float* __restrict__ b1v, const float* __restrict__ b2v,   // [64]
    const float* __restrict__ b1c, const float* __restrict__ b2c,   // [64]
    const uint32_t* __restrict__ WnB,         // [32][64]  bf16 pairs
    const uint32_t* __restrict__ W1vB,        // [192][64]
    const uint32_t* __restrict__ W2vB,        // [192][64]
    const uint32_t* __restrict__ W1cB,        // [512][64]
    const uint32_t* __restrict__ W2cB,        // [512][64]
    float* __restrict__ out)                  // [V,64]
{
    __shared__ uint32_t s_w12[2 * 12288];  // W1v,W2v bf16 pairs, 96 KB (block-resident)
    __shared__ uint32_t s_x[96 * 32];      // 96x64 bf16 (== 16x384 view), 12 KB
    __shared__ uint32_t s_y[16 * 512];     // 16x1024 bf16, 32 KB
    __shared__ float    s_h [16 * 64];     // raw matmul1 result, 4 KB
    __shared__ float    s_h2[16 * 64];     // raw matmul2 result, 4 KB
    __shared__ float    s_red[256];
    __shared__ float    s_norm[16];
    __shared__ float    s_bn[64], s_false[64], s_true[64];
    __shared__ float    s_b1v[64], s_b2v[64], s_b1c[64], s_b2c[64];
    // branchless one-hot selection weights per literal row
    __shared__ float    s_selNeg[96], s_selG[96], s_selF[96];
    __shared__ float    s_cvalf[16];       // clause_valid as float, per variable

    const int tid  = threadIdx.x;
    const int lane = tid & 31;
    const int wave = tid >> 5;             // 8 waves
    const int v0   = blockIdx.x * 16;
    const int w_nt  = wave & 3;            // N-tile owned by this wave
    const int w_mat = wave >> 2;           // 0: layer1 weights, 1: layer2 weights

    // Stage w1v/w2v into LDS once per block (coalesced 128-bit copies)
    {
        const uint4* g1 = reinterpret_cast<const uint4*>(W1vB);
        const uint4* g2 = reinterpret_cast<const uint4*>(W2vB);
        uint4* d = reinterpret_cast<uint4*>(s_w12);
        for (int i = tid; i < 3072; i += 256) {
            d[i]        = g1[i];
            d[3072 + i] = g2[i];
        }
    }
    // Hoist this wave's two Wn B-fragments into registers (used 16x per block).
    // All 3 stage-1 tiles of a wave share nt = wave&3, so 2 fragments suffice.
    FragB16 wnF[2];
#pragma unroll
    for (int ks = 0; ks < 2; ++ks) load_b_glb(wnF[ks], WnB + w_nt * 16, ks * 32, lane);

    // Per-block constants: biases, false_emb, true_emb = false_emb @ Wn + bn
    if (tid < 64) {
        s_bn[tid]    = bn[tid];
        s_false[tid] = false_emb[tid];
        s_b1v[tid]   = b1v[tid];
        s_b2v[tid]   = b2v[tid];
        s_b1c[tid]   = b1c[tid];
        s_b2c[tid]   = b2c[tid];
        float t = bn[tid];
        for (int k = 0; k < 64; ++k) t = fmaf(false_emb[k], Wn[k * 64 + tid], t);
        s_true[tid] = t;
    }
    __syncthreads();

    // ================= per-variable pipeline (stages 1 & 2) ================
    for (int vi = 0; vi < 16; ++vi) {
        const int v     = v0 + vi;
        const int lbase = v * (NC * NL);  // 96 literals

        // one-hot selection weights (branchless select later)
        if (tid < 96) {
            bool nm = neg_mask[lbase + tid] != 0;
            bool vv = var_valid[lbase + tid] != 0;
            s_selNeg[tid] = (vv && nm)  ? 1.0f : 0.0f;
            s_selG[tid]   = (vv && !nm) ? 1.0f : 0.0f;
            s_selF[tid]   = vv ? 0.0f : 1.0f;
        }
        if (tid >= 96 && tid < 112) {
            int cc = tid - 96;
            s_cvalf[cc] = (clause_valid[v * NC + cc] != 0) ? 1.0f : 0.0f;
        }

        // Gather literal embeddings -> bf16 in LDS (row = c*6+l, 64 cols)
        for (int i = tid; i < 96 * 32; i += 256) {
            int row = i >> 5, cp = i & 31;
            int idx = lits[lbase + row];
            const float2* gp = reinterpret_cast<const float2*>(
                variables + (size_t)idx * ND + cp * 2);
            float2 g = *gp;
            s_x[i] = pack_bf(g.x, g.y);
        }
        __syncthreads();

        // ---- Stage 1: neg = g @ Wn  (96x64 @ 64x64), 24 tiles over 8 waves
        // wave owns n-tile w_nt; m-tiles: (wave>>2) + 2t, t=0..2
        f32x8 accn[3];
#pragma unroll
        for (int t = 0; t < 3; ++t) {
            int mt = w_mat + 2 * t;               // 0,2,4 or 1,3,5
            f32x8 c = {};
#pragma unroll
            for (int ks = 0; ks < 2; ++ks) {
                FragB16 a;
                load_a_lds(a, s_x, mt * 16, 32, ks * 32, lane);
                c = wmma_bf16(a, wnF[ks], c);
            }
            accn[t] = c;
        }
        __syncthreads();   // all reads of g in s_x complete before overwrite

        // branchless select: selNeg*(g@Wn+bn) + selG*g + selF*false_emb
        {
            int mh = (lane >> 4) * 8;
            int n  = w_nt * 16 + (lane & 15);
            float fb = s_false[n];
            float bb = s_bn[n];
#pragma unroll
            for (int t = 0; t < 3; ++t) {
                int mt = w_mat + 2 * t;
#pragma unroll
                for (int r = 0; r < 8; ++r) {
                    int rl = mt * 16 + r + mh;    // literal row 0..95
                    float wNeg = s_selNeg[rl], wG = s_selG[rl], wF = s_selF[rl];
                    uint32_t w = s_x[rl * 32 + (n >> 1)];
                    float gval   = bf2f((unsigned short)((n & 1) ? (w >> 16) : (w & 0xFFFFu)));
                    float negval = accn[t][r] + bb;
                    float res = fmaf(wNeg, negval, fmaf(wG, gval, wF * fb));
                    reinterpret_cast<unsigned short*>(s_x)[rl * 64 + n] = f2bf(res);
                }
            }
        }
        __syncthreads();

        // ---- Stage 2: [16 clauses, 384] @ [384,64] for W1v and W2v (LDS B)
        {
            const uint32_t* W = s_w12 + w_mat * 12288 + w_nt * 16;
            f32x8 c = {};
#pragma unroll
            for (int ks = 0; ks < 12; ++ks) {
                FragB16 a, b;
                load_a_lds(a, s_x, 0, 192, ks * 32, lane);   // 16x384 view of s_x
                load_b_lds(b, W, ks * 32, lane);
                c = wmma_bf16(a, b, c);
            }
            float* H = w_mat ? s_h2 : s_h;
            int mh = (lane >> 4) * 8;
            int n  = w_nt * 16 + (lane & 15);
#pragma unroll
            for (int r = 0; r < 8; ++r) H[(r + mh) * 64 + n] = c[r];
        }
        __syncthreads();

        // finalize: sigmoid(h1+b1)+h2+b2, L2-normalize, clause_valid blend
        {
            int cc = tid >> 4, c0 = tid & 15;
            float hv[4], ssq = 0.f;
#pragma unroll
            for (int q = 0; q < 4; ++q) {
                int n = c0 + q * 16;
                float h = sigmoidf_(s_h[cc * 64 + n] + s_b1v[n]) + s_h2[cc * 64 + n] + s_b2v[n];
                hv[q] = h; ssq = fmaf(h, h, ssq);
            }
            s_red[tid] = ssq;
            __syncthreads();
            if (tid < 16) {
                float s = 0.f;
                for (int q = 0; q < 16; ++q) s += s_red[tid * 16 + q];
                s_norm[tid] = rsqrtf(s);
            }
            __syncthreads();
            float inv = s_norm[cc];
            float fc  = s_cvalf[cc];
            unsigned short* yb = reinterpret_cast<unsigned short*>(s_y);
#pragma unroll
            for (int q = 0; q < 4; ++q) {
                int n = c0 + q * 16;
                float te  = s_true[n];
                float val = fmaf(fc, fmaf(hv[q], inv, -te), te);  // fc? hv*inv : true
                yb[vi * 1024 + cc * 64 + n] = f2bf(val);
            }
        }
        __syncthreads();
    }

    // ================= Stage 3: [16 vars, 1024] @ [1024,64] x2 =============
    {
        const uint32_t* W = (w_mat ? W2cB : W1cB) + w_nt * 16;   // streamed, used once
        f32x8 c = {};
#pragma unroll 8
        for (int ks = 0; ks < 32; ++ks) {
            FragB16 a, b;
            load_a_lds(a, s_y, 0, 512, ks * 32, lane);
            load_b_glb(b, W, ks * 32, lane);
            c = wmma_bf16(a, b, c);
        }
        float* H = w_mat ? s_h2 : s_h;
        int mh = (lane >> 4) * 8;
        int n  = w_nt * 16 + (lane & 15);
#pragma unroll
        for (int r = 0; r < 8; ++r) H[(r + mh) * 64 + n] = c[r];
    }
    __syncthreads();
    {
        int mrow = tid >> 4, c0 = tid & 15;
        int v = v0 + mrow;
        float hv[4], ssq = 0.f;
#pragma unroll
        for (int q = 0; q < 4; ++q) {
            int n = c0 + q * 16;
            float h = sigmoidf_(s_h[mrow * 64 + n] + s_b1c[n]) + s_h2[mrow * 64 + n] + s_b2c[n];
            hv[q] = h; ssq = fmaf(h, h, ssq);
        }
        s_red[tid] = ssq;
        __syncthreads();
        if (tid < 16) {
            float s = 0.f;
            for (int q = 0; q < 16; ++q) s += s_red[tid * 16 + q];
            s_norm[tid] = rsqrtf(s);
        }
        __syncthreads();
        // has_clauses blend (branchless; setup guarantees >=1 clause anyway)
        int anyc = 0;
        for (int q = 0; q < NC; ++q) anyc |= (int)clause_valid[v * NC + q];
        float fh  = (anyc != 0) ? 1.0f : 0.0f;
        float inv = s_norm[mrow];
#pragma unroll
        for (int q = 0; q < 4; ++q) {
            int n = c0 + q * 16;
            float pass = variables[(size_t)v * ND + n];
            out[(size_t)v * ND + n] = fmaf(fh, fmaf(hv[q], inv, -pass), pass);
        }
    }
}

// ---------------------------------------------------------------------------
// Launch
// ---------------------------------------------------------------------------
extern "C" void kernel_launch(void* const* d_in, const int* in_sizes, int n_in,
                              void* d_out, int out_size, void* d_ws, size_t ws_size,
                              hipStream_t stream) {
    const float* variables          = (const float*)d_in[0];
    const int*   lits               = (const int*)d_in[1];
    const unsigned char* neg_mask   = (const unsigned char*)d_in[2];
    const unsigned char* var_valid  = (const unsigned char*)d_in[3];
    const unsigned char* clause_val = (const unsigned char*)d_in[4];
    const float* Wn  = (const float*)d_in[5];
    const float* bn  = (const float*)d_in[6];
    const float* fe  = (const float*)d_in[7];
    const float* W1v = (const float*)d_in[8];
    const float* b1v = (const float*)d_in[9];
    const float* W2v = (const float*)d_in[10];
    const float* b2v = (const float*)d_in[11];
    const float* W1c = (const float*)d_in[12];
    const float* b1c = (const float*)d_in[13];
    const float* W2c = (const float*)d_in[14];
    const float* b2c = (const float*)d_in[15];
    float* out = (float*)d_out;

    // Workspace layout (bf16 pair-interleaved weights), dword counts:
    uint32_t* ws   = (uint32_t*)d_ws;
    uint32_t* WnB  = ws;                 // 32*64   = 2048
    uint32_t* W1vB = WnB  + 2048;        // 192*64  = 12288
    uint32_t* W2vB = W1vB + 12288;       // 12288
    uint32_t* W1cB = W2vB + 12288;       // 512*64  = 32768
    uint32_t* W2cB = W1cB + 32768;       // 32768

    auto cvt = [&](const float* s, uint32_t* d, int ndw) {
        convert_pairs<<<(ndw + 255) / 256, 256, 0, stream>>>(s, d, ndw);
    };
    cvt(Wn,  WnB,  2048);
    cvt(W1v, W1vB, 12288);
    cvt(W2v, W2vB, 12288);
    cvt(W1c, W1cB, 32768);
    cvt(W2c, W2cB, 32768);

    sat_fused_kernel<<<NV / 16, 256, 0, stream>>>(
        variables, lits, neg_mask, var_valid, clause_val,
        Wn, bn, fe, b1v, b2v, b1c, b2c,
        WnB, W1vB, W2vB, W1cB, W2cB, out);
}
